// SlotLatentDynamicsModel_86887188398405
// MI455X (gfx1250) — matrix operand order repaired and checked by reading, not compiled, not run on hardware
//
#include <hip/hip_runtime.h>
#include <math.h>

// ---------------------------------------------------------------------------
// SlotLatentDynamicsModel on gfx1250 (MI455X).
// Heavy GEMMs (dynamics transformer + decoder, ~690 GFLOPs) run through
// v_wmma_f32_16x16x32_bf16 with f32 accumulation. GEMM tiles are staged
// global->LDS by the Tensor Data Mover (tensor_load_to_lds, TENSORcnt),
// so the compute waves never touch the tiles with vector loads.
// LN / softmax / GRU scan stay in f32.
// ---------------------------------------------------------------------------

typedef __bf16 bf16;
typedef __attribute__((ext_vector_type(16))) __bf16 v16bf;
typedef __attribute__((ext_vector_type(8)))  float  v8f;
typedef __attribute__((ext_vector_type(4)))  float  f32x4;
typedef __attribute__((ext_vector_type(4)))  unsigned int u32x4;
typedef __attribute__((ext_vector_type(8)))  int    i32x8;
typedef __attribute__((ext_vector_type(4)))  int    i32x4;

#define B_   32
#define T_   256
#define C_   8
#define HW_  64      // H*W tokens per frame
#define K_   8       // slots
#define D_   64      // slot dim
#define MD_  512
#define NH_  8
#define HD_  64
#define FF_  2048
#define L_   12
#define AD_  32

__device__ __forceinline__ float bf2f(bf16 x) {
    unsigned short h = __builtin_bit_cast(unsigned short, x);
    unsigned u = ((unsigned)h) << 16;
    return __builtin_bit_cast(float, u);
}
__device__ __forceinline__ bf16 f2bf(float x) {
    unsigned u = __builtin_bit_cast(unsigned, x);
    unsigned r = u + 0x7fffu + ((u >> 16) & 1u);   // round-to-nearest-even
    unsigned short h = (unsigned short)(r >> 16);
    return __builtin_bit_cast(bf16, h);
}
__device__ __forceinline__ float gelu_exact(float x) {
    return 0.5f * x * (1.0f + erff(x * 0.70710678118654752f));
}

// ---------------------------------------------------------------------------
// Tensor Data Mover: 2-D tile DMA global -> LDS (bf16 elements).
// D# packing per CDNA5 ISA 8.3/8.4:
//   group0: [1:0]=count=1, [63:32]=lds_addr, [120:64]=global byte addr,
//           [127:126]=type=2
//   group1: [17:16]=data_size (1 => 2 bytes), [79:48]=tensor_dim0,
//           [111:80]=tensor_dim1, [127:112]=tile_dim0, [143:128]=tile_dim1,
//           [207:160]=tensor_dim0_stride (elements)
// dims are the remaining extent from the tile origin -> OOB reads zero-fill,
// giving free edge padding. Tile is written X-major, matching the LDS layout.
// ---------------------------------------------------------------------------
__device__ __forceinline__ void tdm_load_2d(unsigned lds_off, unsigned long gaddr,
                                            int dim0, int dim1, int stride0,
                                            int tile0, int tile1)
{
    u32x4 g0;
    g0.x = 1u;                                           // count=1 (valid user D#)
    g0.y = lds_off;                                      // LDS byte address
    g0.z = (unsigned)(gaddr & 0xffffffffu);              // global addr [31:0]
    g0.w = (unsigned)((gaddr >> 32) & 0x01ffffffu) | (2u << 30);  // [56:32] | type=2
    i32x8 g1;
    g1[0] = (1 << 16);                                   // data_size = 2 bytes
    g1[1] = (int)(((unsigned)dim0 & 0xffffu) << 16);     // tensor_dim0 [15:0]
    g1[2] = (int)((((unsigned)dim0 >> 16) & 0xffffu) |
                  (((unsigned)dim1 & 0xffffu) << 16));   // dim0 hi | dim1 lo
    g1[3] = (int)((((unsigned)dim1 >> 16) & 0xffffu) |
                  (((unsigned)tile0 & 0xffffu) << 16));  // dim1 hi | tile_dim0
    g1[4] = (int)((unsigned)tile1 & 0xffffu);            // tile_dim1 (tile_dim2=0)
    g1[5] = stride0;                                     // dim0 stride [31:0]
    g1[6] = 0;                                           // stride hi | dim1 stride
    g1[7] = 0;
    i32x4 z4 = {0, 0, 0, 0};
#if __clang_major__ >= 23
    i32x8 z8 = {0, 0, 0, 0, 0, 0, 0, 0};
    __builtin_amdgcn_tensor_load_to_lds(g0, g1, z4, z4, z8, 0);
#else
    __builtin_amdgcn_tensor_load_to_lds(g0, g1, z4, z4, 0);
#endif
}

// ---------------------------------------------------------------------------
// f32 -> bf16 bulk convert
// ---------------------------------------------------------------------------
__global__ void convert_f32_to_bf16(const float* __restrict__ src,
                                    bf16* __restrict__ dst, long n) {
    long i = (long)blockIdx.x * 256 + threadIdx.x;
    long stride = (long)gridDim.x * 256;
    for (; i < n; i += stride) dst[i] = f2bf(src[i]);
}

// ---------------------------------------------------------------------------
// Generic batched bf16 GEMM: C[z] = A[z] @ B[z] (+bias) (+Cin) (gelu?)
//   A: [M,K] bf16 row-major, B: [K,N] bf16 row-major (natural weight layout)
//   Tile 64x128, 8 waves of 2x2 16x16 WMMA tiles, BK=32, double-buffered LDS,
//   tiles staged by TDM (wave 0 issues, all waves sync on TENSORcnt+barrier).
// Fragment layouts per CDNA5 ISA 7.12.2:
//   A frag: lane = M row (mod 16), K chunks {0..7,16..23} / {8..15,24..31}
//   B frag: lane = N col (mod 16), same K chunks as A (strided LDS reads)
// ---------------------------------------------------------------------------
#define BM 64
#define BN 128
#define BK 32

union Frag { v16bf v; f32x4 f[2]; };

__global__ __launch_bounds__(256) void gemm_bf16_kernel(
    const bf16* __restrict__ A, const bf16* __restrict__ Bm,
    const float* __restrict__ bias, const float* __restrict__ Cin,
    float* __restrict__ Cout, bf16* __restrict__ CoutB,
    int M, int N, int K, long sA, long sB, long sC, int gelu)
{
    __shared__ __align__(16) bf16 lsA[2][BM][BK];
    __shared__ __align__(16) bf16 lsB[2][BK][BN];

    const int z = blockIdx.z;
    A  += (long)z * sA;
    Bm += (long)z * sB;
    const long co = (long)z * sC;

    const int m0 = blockIdx.y * BM, n0 = blockIdx.x * BN;
    const int tid = threadIdx.x, lane = tid & 31;
    const int w = tid >> 5, wr = w >> 2, wc = w & 3;

    v8f acc[2][2] = {};

    // wave 0 drives the Tensor Data Mover; tiles land X-major in LDS
    auto issue = [&](int buf, int k0) {
        if (w == 0) {
            tdm_load_2d((unsigned)(size_t)&lsA[buf][0][0],
                        (unsigned long)(size_t)(A + (long)m0 * K + k0),
                        K - k0, M - m0, K, BK, BM);
            tdm_load_2d((unsigned)(size_t)&lsB[buf][0][0],
                        (unsigned long)(size_t)(Bm + (long)k0 * N + n0),
                        N - n0, K - k0, N, BN, BK);
        }
    };

    issue(0, 0);
    __builtin_amdgcn_s_wait_tensorcnt(0);
    __syncthreads();

    int buf = 0;
    const int lm = lane & 15;
    const int kb = (lane < 16) ? 0 : 8;
    for (int k0 = 0; k0 < K; k0 += BK) {
        if (k0 + BK < K) issue(buf ^ 1, k0 + BK);   // DMA next tile while computing

        Frag af[2], bfr[2];
#pragma unroll
        for (int mi = 0; mi < 2; ++mi) {
            const bf16* p = &lsA[buf][wr * 32 + mi * 16 + lm][0];
            af[mi].f[0] = *(const f32x4*)(p + kb);
            af[mi].f[1] = *(const f32x4*)(p + kb + 16);
        }
#pragma unroll
        for (int ni = 0; ni < 2; ++ni) {
            const bf16* p = &lsB[buf][0][wc * 32 + ni * 16 + lm];
#pragma unroll
            for (int s = 0; s < 8; ++s) {
                bfr[ni].v[s]     = p[(kb + s) * BN];
                bfr[ni].v[s + 8] = p[(kb + 16 + s) * BN];
            }
        }
#pragma unroll
        for (int mi = 0; mi < 2; ++mi)
#pragma unroll
            for (int ni = 0; ni < 2; ++ni)
                acc[mi][ni] = __builtin_amdgcn_wmma_f32_16x16x32_bf16(
                    false, af[mi].v, false, bfr[ni].v,
                    (short)0, acc[mi][ni], false, false);

        __builtin_amdgcn_s_wait_tensorcnt(0);       // wave 0: my DMA landed
        __syncthreads();                            // everyone: LDS visible
        buf ^= 1;
    }

    const int hi = lane >> 4;                 // lanes 16-31 cover M rows +8
#pragma unroll
    for (int mi = 0; mi < 2; ++mi) {
#pragma unroll
        for (int ni = 0; ni < 2; ++ni) {
            int n = n0 + wc * 32 + ni * 16 + lm;
            if (n >= N) continue;
            int mbase = m0 + wr * 32 + mi * 16 + hi * 8;
#pragma unroll
            for (int r = 0; r < 8; ++r) {
                int m = mbase + r;
                if (m >= M) continue;
                float v = acc[mi][ni][r];
                if (bias) v += bias[n];
                long o = co + (long)m * N + n;
                if (Cin)  v += Cin[o];
                if (gelu) v = gelu_exact(v);
                if (Cout)  Cout[o]  = v;
                if (CoutB) CoutB[o] = f2bf(v);
            }
        }
    }
}

// ---------------------------------------------------------------------------
// Stage 1: tokens = x@lat_proj + b + pos; LN; k = inp@wk; v = inp@wv (bf16)
// one block per (b,t) frame, 64 threads = 64 tokens
// ---------------------------------------------------------------------------
__global__ __launch_bounds__(64) void stage1_kernel(
    const float* __restrict__ latents, const float* __restrict__ lpw,
    const float* __restrict__ lpb, const float* __restrict__ pos,
    const float* __restrict__ g, const float* __restrict__ bb,
    const float* __restrict__ wk, const float* __restrict__ wv,
    bf16* __restrict__ kout, bf16* __restrict__ vout)
{
    __shared__ float swk[D_ * D_];
    __shared__ float swv[D_ * D_];
    const int bt = blockIdx.x, n = threadIdx.x;
    for (int i = n; i < D_ * D_; i += 64) { swk[i] = wk[i]; swv[i] = wv[i]; }
    __syncthreads();

    const float* lp = latents + (long)bt * (C_ * HW_);
    float x[C_];
#pragma unroll
    for (int c = 0; c < C_; ++c) x[c] = lp[c * HW_ + n];

    float tok[D_];
    for (int d = 0; d < D_; ++d) {
        float s = lpb[d] + pos[n * D_ + d];
#pragma unroll
        for (int c = 0; c < C_; ++c) s += x[c] * lpw[c * D_ + d];
        tok[d] = s;
    }
    float m = 0.f;
    for (int d = 0; d < D_; ++d) m += tok[d];
    m *= (1.0f / D_);
    float var = 0.f;
    for (int d = 0; d < D_; ++d) { float dd = tok[d] - m; var += dd * dd; }
    var *= (1.0f / D_);
    float inv = rsqrtf(var + 1e-5f);
    for (int d = 0; d < D_; ++d) tok[d] = (tok[d] - m) * inv * g[d] + bb[d];

    const long base = (long)bt * (HW_ * D_) + (long)n * D_;
    for (int d = 0; d < D_; ++d) {
        float ks = 0.f, vs = 0.f;
        for (int e = 0; e < D_; ++e) {
            ks += tok[e] * swk[e * D_ + d];
            vs += tok[e] * swv[e * D_ + d];
        }
        kout[base + d] = f2bf(ks);
        vout[base + d] = f2bf(vs);
    }
}

// ---------------------------------------------------------------------------
// Stage 2: slot-attention scan over T (sequential). One block per batch.
// ---------------------------------------------------------------------------
__global__ __launch_bounds__(256) void slot_scan_kernel(
    const float* __restrict__ prev_slots,
    const bf16* __restrict__ kbuf, const bf16* __restrict__ vbuf,
    const float* __restrict__ wq,
    const float* __restrict__ wih, const float* __restrict__ whh,
    const float* __restrict__ bih, const float* __restrict__ bhh,
    const float* __restrict__ w1, const float* __restrict__ b1,
    const float* __restrict__ w2, const float* __restrict__ b2,
    const float* __restrict__ lng, const float* __restrict__ lnb,
    bf16* __restrict__ seq_bf, float* __restrict__ slots_final)
{
    const int b = blockIdx.x, tid = threadIdx.x;
    __shared__ float slots[K_ * D_], prevs[K_ * D_], q[K_ * D_];
    __shared__ float att[K_ * HW_], upd[K_ * D_];
    __shared__ float gi[K_ * 3 * D_], gh[K_ * 3 * D_];
    __shared__ float hb[K_ * D_], act[K_ * 4 * D_];
    __shared__ float lmean[K_], linv[K_];

    for (int i = tid; i < K_ * D_; i += 256) slots[i] = prev_slots[b * K_ * D_ + i];
    __syncthreads();

    for (int t = 0; t < T_; ++t) {
        const bf16* kt = kbuf + ((long)b * T_ + t) * (HW_ * D_);
        const bf16* vt = vbuf + ((long)b * T_ + t) * (HW_ * D_);
        for (int i = tid; i < K_ * D_; i += 256) slots[i] *= 0.95f;  // decay prior
        __syncthreads();

        for (int iter = 0; iter < 3; ++iter) {
            for (int i = tid; i < K_ * D_; i += 256) prevs[i] = slots[i];
            __syncthreads();
            // q = slots @ wq
            for (int o = tid; o < K_ * D_; o += 256) {
                int k = o >> 6, d = o & 63;
                float s = 0.f;
                for (int e = 0; e < D_; ++e) s += prevs[k * D_ + e] * wq[e * D_ + d];
                q[o] = s;
            }
            __syncthreads();
            // logits[k][n] (scaled)
            for (int o = tid; o < K_ * HW_; o += 256) {
                int k = o >> 6, n = o & 63;
                float s = 0.f;
                for (int d = 0; d < D_; ++d) s += q[k * D_ + d] * bf2f(kt[n * D_ + d]);
                att[o] = s * 0.125f;
            }
            __syncthreads();
            // softmax over slot axis per token n
            if (tid < HW_) {
                int n = tid;
                float mx = -3.0e38f;
                for (int k = 0; k < K_; ++k) mx = fmaxf(mx, att[k * HW_ + n]);
                float sm = 0.f;
                for (int k = 0; k < K_; ++k) {
                    float e = __expf(att[k * HW_ + n] - mx);
                    att[k * HW_ + n] = e; sm += e;
                }
                float inv = 1.f / sm;
                for (int k = 0; k < K_; ++k) att[k * HW_ + n] *= inv;
            }
            __syncthreads();
            // upd = attn @ v
            for (int o = tid; o < K_ * D_; o += 256) {
                int k = o >> 6, d = o & 63;
                float s = 0.f;
                for (int n = 0; n < HW_; ++n) s += att[k * HW_ + n] * bf2f(vt[n * D_ + d]);
                upd[o] = s;
            }
            __syncthreads();
            // GRU gates
            for (int o = tid; o < K_ * 3 * D_; o += 256) {
                int k = o / (3 * D_), j = o % (3 * D_);
                float s = bih[j], s2 = bhh[j];
                for (int e = 0; e < D_; ++e) {
                    s  += upd[k * D_ + e]   * wih[e * 3 * D_ + j];
                    s2 += prevs[k * D_ + e] * whh[e * 3 * D_ + j];
                }
                gi[o] = s; gh[o] = s2;
            }
            __syncthreads();
            for (int o = tid; o < K_ * D_; o += 256) {
                int k = o >> 6, d = o & 63;
                float r  = 1.f / (1.f + __expf(-(gi[k * 3 * D_ + d] + gh[k * 3 * D_ + d])));
                float zg = 1.f / (1.f + __expf(-(gi[k * 3 * D_ + D_ + d] + gh[k * 3 * D_ + D_ + d])));
                float nn = tanhf(gi[k * 3 * D_ + 2 * D_ + d] + r * gh[k * 3 * D_ + 2 * D_ + d]);
                slots[o] = (1.f - zg) * nn + zg * prevs[o];
            }
            __syncthreads();
            // slot LN
            if (tid < K_) {
                float m = 0.f;
                for (int d = 0; d < D_; ++d) m += slots[tid * D_ + d];
                m *= (1.0f / D_);
                float v = 0.f;
                for (int d = 0; d < D_; ++d) { float dd = slots[tid * D_ + d] - m; v += dd * dd; }
                v *= (1.0f / D_);
                lmean[tid] = m; linv[tid] = rsqrtf(v + 1e-5f);
            }
            __syncthreads();
            for (int o = tid; o < K_ * D_; o += 256) {
                int k = o >> 6, d = o & 63;
                hb[o] = (slots[o] - lmean[k]) * linv[k] * lng[d] + lnb[d];
            }
            __syncthreads();
            // MLP
            for (int o = tid; o < K_ * 4 * D_; o += 256) {
                int k = o / (4 * D_), j = o % (4 * D_);
                float s = b1[j];
                for (int e = 0; e < D_; ++e) s += hb[k * D_ + e] * w1[e * 4 * D_ + j];
                act[o] = gelu_exact(s);
            }
            __syncthreads();
            for (int o = tid; o < K_ * D_; o += 256) {
                int k = o >> 6, d = o & 63;
                float s = b2[d];
                for (int j = 0; j < 4 * D_; ++j) s += act[k * 4 * D_ + j] * w2[j * D_ + d];
                slots[o] += s;
            }
            __syncthreads();
        }
        for (int i = tid; i < K_ * D_; i += 256)
            seq_bf[((long)b * T_ + t) * (K_ * D_) + i] = f2bf(slots[i]);
        __syncthreads();
    }
    for (int i = tid; i < K_ * D_; i += 256) slots_final[b * K_ * D_ + i] = slots[i];
}

// ---------------------------------------------------------------------------
// LayerNorm over rows of 512; optional bf16 and/or f32 out, optional scale ptr
// ---------------------------------------------------------------------------
__global__ __launch_bounds__(256) void layernorm_kernel(
    const float* __restrict__ x, const float* __restrict__ g,
    const float* __restrict__ b, bf16* __restrict__ outb,
    float* __restrict__ outf, const float* __restrict__ scale_ptr, int rows)
{
    __shared__ float red[256];
    const int row = blockIdx.x, tid = threadIdx.x;
    const float* xr = x + (long)row * MD_;
    float v0 = xr[tid], v1 = xr[tid + 256];
    red[tid] = v0 + v1;
    __syncthreads();
    for (int o = 128; o > 0; o >>= 1) { if (tid < o) red[tid] += red[tid + o]; __syncthreads(); }
    float mean = red[0] * (1.0f / MD_);
    __syncthreads();
    float d0 = v0 - mean, d1 = v1 - mean;
    red[tid] = d0 * d0 + d1 * d1;
    __syncthreads();
    for (int o = 128; o > 0; o >>= 1) { if (tid < o) red[tid] += red[tid + o]; __syncthreads(); }
    float inv = rsqrtf(red[0] * (1.0f / MD_) + 1e-5f);
    float sc = scale_ptr ? scale_ptr[0] : 1.0f;
    float o0 = (d0 * inv * g[tid]       + b[tid])       * sc;
    float o1 = (d1 * inv * g[tid + 256] + b[tid + 256]) * sc;
    long base = (long)row * MD_;
    if (outb) { outb[base + tid] = f2bf(o0); outb[base + tid + 256] = f2bf(o1); }
    if (outf) { outf[base + tid] = o0;        outf[base + tid + 256] = o1; }
}

// ---------------------------------------------------------------------------
// Causal softmax over T=256 (WIN=256 >= T -> pure causal), scaled by 1/8
// ---------------------------------------------------------------------------
__global__ __launch_bounds__(256) void softmax_kernel(
    const float* __restrict__ s, bf16* __restrict__ a)
{
    __shared__ float red[256];
    const long row = blockIdx.x;          // (b*NH+h)*T + i
    const int  j = threadIdx.x;
    const int  i = (int)(row & (T_ - 1));
    float x = (j <= i) ? s[row * T_ + j] * 0.125f : -3.0e38f;
    red[j] = x;
    __syncthreads();
    for (int o = 128; o > 0; o >>= 1) { if (j < o) red[j] = fmaxf(red[j], red[j + o]); __syncthreads(); }
    float mx = red[0];
    __syncthreads();
    float e = (j <= i) ? __expf(x - mx) : 0.f;
    red[j] = e;
    __syncthreads();
    for (int o = 128; o > 0; o >>= 1) { if (j < o) red[j] += red[j + o]; __syncthreads(); }
    float inv = 1.f / red[0];
    a[row * T_ + j] = f2bf(e * inv);
}

// ---------------------------------------------------------------------------
// Repack qkv [B,T,3*512] -> q[B,NH,T,HD], kT[B,NH,HD,T], v[B,NH,T,HD]
// ---------------------------------------------------------------------------
__global__ void repack_qkv_kernel(const bf16* __restrict__ qkv,
                                  bf16* __restrict__ q, bf16* __restrict__ kt,
                                  bf16* __restrict__ v)
{
    long gid = (long)blockIdx.x * 256 + threadIdx.x;
    if (gid >= (long)B_ * T_ * MD_) return;
    int d = (int)(gid & 63);
    int h = (int)((gid >> 6) & 7);
    long bt = gid >> 9;
    int t = (int)(bt % T_), b = (int)(bt / T_);
    long base = bt * (3 * MD_);
    long bh = (long)b * NH_ + h;
    q [(bh * T_ + t) * HD_ + d]  = qkv[base + h * HD_ + d];
    kt[(bh * HD_ + d) * T_ + t]  = qkv[base + MD_ + h * HD_ + d];
    v [(bh * T_ + t) * HD_ + d]  = qkv[base + 2 * MD_ + h * HD_ + d];
}

// attn out [B,NH,T,HD] -> [B,T,NH*HD]
__global__ void repack_o_kernel(const bf16* __restrict__ o, bf16* __restrict__ o2)
{
    long gid = (long)blockIdx.x * 256 + threadIdx.x;
    if (gid >= (long)B_ * T_ * MD_) return;
    int d = (int)(gid & 63);
    int h = (int)((gid >> 6) & 7);
    long bt = gid >> 9;
    int t = (int)(bt % T_), b = (int)(bt / T_);
    o2[gid] = o[(((long)b * NH_ + h) * T_ + t) * HD_ + d];
}

// ---------------------------------------------------------------------------
// Host side
// ---------------------------------------------------------------------------
enum {
    IN_LATENTS = 0, IN_ACTIONS, IN_PREV_SLOTS,
    P_LAT_PROJ_W, P_LAT_PROJ_B, P_POS_EMBED, P_ATTN_LN_G, P_ATTN_LN_B,
    P_WQ, P_WK, P_WV, P_GRU_WIH, P_GRU_WHH, P_GRU_BIH, P_GRU_BHH,
    P_MLP_W1, P_MLP_B1, P_MLP_W2, P_MLP_B2, P_SLOT_LN_G, P_SLOT_LN_B,
    P_DYN_IN_W, P_DYN_IN_B, P_DYN_ACT_W, P_DYN_ACT_B,
    P_DYN_LN1_G, P_DYN_LN1_B, P_DYN_WQKV, P_DYN_BQKV, P_DYN_WO, P_DYN_BO,
    P_DYN_LN2_G, P_DYN_LN2_B, P_DYN_FFN_W1, P_DYN_FFN_B1, P_DYN_FFN_W2, P_DYN_FFN_B2,
    P_DYN_LNF_G, P_DYN_LNF_B, P_DYN_OUT_W, P_DYN_OUT_B,
    P_DEC_W1, P_DEC_B1, P_DEC_W2, P_DEC_B2, P_OUT_LN_G, P_OUT_LN_B, P_LATENT_SCALE
};

extern "C" void kernel_launch(void* const* d_in, const int* in_sizes, int n_in,
                              void* d_out, int out_size, void* d_ws, size_t ws_size,
                              hipStream_t stream)
{
    (void)in_sizes; (void)n_in; (void)out_size; (void)ws_size;
    const int ROWS = B_ * T_;                    // 8192
    const long OUT_D = (long)ROWS * MD_;         // 4194304

    const float* latents    = (const float*)d_in[IN_LATENTS];
    const float* actions    = (const float*)d_in[IN_ACTIONS];
    const float* prev_slots = (const float*)d_in[IN_PREV_SLOTS];
    auto P = [&](int i) { return (const float*)d_in[i]; };

    // ---- workspace carve-up ----
    char* ws = (char*)d_ws;
    size_t off = 0;
    auto alloc = [&](size_t bytes) -> void* {
        off = (off + 255) & ~(size_t)255;
        void* p = ws + off; off += bytes; return p;
    };
    bf16* w_in  = (bf16*)alloc((size_t)MD_ * MD_ * 2);
    bf16* w_act = (bf16*)alloc((size_t)AD_ * MD_ * 2);
    bf16* w_qkv = (bf16*)alloc((size_t)L_ * MD_ * 3 * MD_ * 2);
    bf16* w_o   = (bf16*)alloc((size_t)L_ * MD_ * MD_ * 2);
    bf16* w_f1  = (bf16*)alloc((size_t)L_ * MD_ * FF_ * 2);
    bf16* w_f2  = (bf16*)alloc((size_t)L_ * FF_ * MD_ * 2);
    bf16* w_out = (bf16*)alloc((size_t)MD_ * MD_ * 2);
    bf16* w_d1  = (bf16*)alloc((size_t)MD_ * MD_ * 2);
    bf16* w_d2  = (bf16*)alloc((size_t)MD_ * MD_ * 2);
    bf16* act_bf = (bf16*)alloc((size_t)ROWS * AD_ * 2);
    bf16* kbuf  = (bf16*)alloc((size_t)ROWS * HW_ * D_ * 2);
    bf16* vbuf  = (bf16*)alloc((size_t)ROWS * HW_ * D_ * 2);
    bf16* seq_bf = (bf16*)alloc((size_t)ROWS * MD_ * 2);
    float* h    = (float*)alloc((size_t)ROWS * MD_ * 4);
    bf16* y_bf  = (bf16*)alloc((size_t)ROWS * MD_ * 2);
    bf16* qkv_bf = (bf16*)alloc((size_t)ROWS * 3 * MD_ * 2);
    bf16* q_bf  = (bf16*)alloc((size_t)ROWS * MD_ * 2);
    bf16* kt_bf = (bf16*)alloc((size_t)ROWS * MD_ * 2);
    bf16* v_bf  = (bf16*)alloc((size_t)ROWS * MD_ * 2);
    float* s_f  = (float*)alloc((size_t)B_ * NH_ * T_ * T_ * 4);
    bf16* a_bf  = (bf16*)alloc((size_t)B_ * NH_ * T_ * T_ * 2);
    bf16* o_bf  = (bf16*)alloc((size_t)ROWS * MD_ * 2);
    bf16* o2_bf = (bf16*)alloc((size_t)ROWS * MD_ * 2);
    bf16* f_bf  = (bf16*)alloc((size_t)ROWS * FF_ * 2);
    bf16* pred_bf = (bf16*)alloc((size_t)ROWS * MD_ * 2);
    bf16* d1_bf = (bf16*)alloc((size_t)ROWS * MD_ * 2);
    float* d2_f = (float*)alloc((size_t)ROWS * MD_ * 4);

    // ---- helpers ----
    auto conv = [&](const float* src, bf16* dst, long n) {
        long blocks = (n + 255) / 256; if (blocks > 8192) blocks = 8192;
        convert_f32_to_bf16<<<(int)blocks, 256, 0, stream>>>(src, dst, n);
    };
    auto gemm = [&](const bf16* A, const bf16* Bm, const float* bias,
                    const float* Cin, float* Cout, bf16* CoutB,
                    int M, int N, int Kd, int Z, long sA, long sB, long sC, int gelu) {
        dim3 g((N + BN - 1) / BN, (M + BM - 1) / BM, Z);
        gemm_bf16_kernel<<<g, 256, 0, stream>>>(A, Bm, bias, Cin, Cout, CoutB,
                                                M, N, Kd, sA, sB, sC, gelu);
    };
    auto ln = [&](const float* x, const float* g, const float* b,
                  bf16* ob, float* of, const float* sc, int rows) {
        layernorm_kernel<<<rows, 256, 0, stream>>>(x, g, b, ob, of, sc, rows);
    };

    // ---- weight / input conversion to bf16 ----
    conv(P(P_DYN_IN_W),  w_in,  (long)MD_ * MD_);
    conv(P(P_DYN_ACT_W), w_act, (long)AD_ * MD_);
    conv(P(P_DYN_WQKV),  w_qkv, (long)L_ * MD_ * 3 * MD_);
    conv(P(P_DYN_WO),    w_o,   (long)L_ * MD_ * MD_);
    conv(P(P_DYN_FFN_W1), w_f1, (long)L_ * MD_ * FF_);
    conv(P(P_DYN_FFN_W2), w_f2, (long)L_ * FF_ * MD_);
    conv(P(P_DYN_OUT_W), w_out, (long)MD_ * MD_);
    conv(P(P_DEC_W1),    w_d1,  (long)MD_ * MD_);
    conv(P(P_DEC_W2),    w_d2,  (long)MD_ * MD_);
    conv(actions,        act_bf, (long)ROWS * AD_);

    // ---- stage 1: per-frame tokens -> k/v ----
    stage1_kernel<<<ROWS, 64, 0, stream>>>(
        latents, P(P_LAT_PROJ_W), P(P_LAT_PROJ_B), P(P_POS_EMBED),
        P(P_ATTN_LN_G), P(P_ATTN_LN_B), P(P_WK), P(P_WV), kbuf, vbuf);

    // ---- stage 2: slot attention scan (writes seq + slots_final) ----
    slot_scan_kernel<<<B_, 256, 0, stream>>>(
        prev_slots, kbuf, vbuf, P(P_WQ),
        P(P_GRU_WIH), P(P_GRU_WHH), P(P_GRU_BIH), P(P_GRU_BHH),
        P(P_MLP_W1), P(P_MLP_B1), P(P_MLP_W2), P(P_MLP_B2),
        P(P_SLOT_LN_G), P(P_SLOT_LN_B),
        seq_bf, (float*)d_out + OUT_D);

    // ---- dynamics input: h = seq@W_in + b + act@W_act + b_act ----
    gemm(seq_bf, w_in,  P(P_DYN_IN_B),  nullptr, h, nullptr, ROWS, MD_, MD_, 1, 0, 0, 0, 0);
    gemm(act_bf, w_act, P(P_DYN_ACT_B), h,       h, nullptr, ROWS, MD_, AD_, 1, 0, 0, 0, 0);

    // ---- 12 transformer layers ----
    const int BH = B_ * NH_;                     // 256 attention batches
    for (int l = 0; l < L_; ++l) {
        const bf16*  wqkv_l = w_qkv + (size_t)l * MD_ * 3 * MD_;
        const bf16*  wo_l   = w_o   + (size_t)l * MD_ * MD_;
        const bf16*  wf1_l  = w_f1  + (size_t)l * MD_ * FF_;
        const bf16*  wf2_l  = w_f2  + (size_t)l * FF_ * MD_;
        const float* ln1g = P(P_DYN_LN1_G) + l * MD_;
        const float* ln1b = P(P_DYN_LN1_B) + l * MD_;
        const float* ln2g = P(P_DYN_LN2_G) + l * MD_;
        const float* ln2b = P(P_DYN_LN2_B) + l * MD_;
        const float* bqkv = P(P_DYN_BQKV) + l * 3 * MD_;
        const float* bo   = P(P_DYN_BO)   + l * MD_;
        const float* fb1  = P(P_DYN_FFN_B1) + l * FF_;
        const float* fb2  = P(P_DYN_FFN_B2) + l * MD_;

        ln(h, ln1g, ln1b, y_bf, nullptr, nullptr, ROWS);
        gemm(y_bf, wqkv_l, bqkv, nullptr, nullptr, qkv_bf, ROWS, 3 * MD_, MD_, 1, 0, 0, 0, 0);
        repack_qkv_kernel<<<(ROWS * MD_) / 256, 256, 0, stream>>>(qkv_bf, q_bf, kt_bf, v_bf);
        // scores: per (b,h): [T,HD] @ [HD,T]
        gemm(q_bf, kt_bf, nullptr, nullptr, s_f, nullptr,
             T_, T_, HD_, BH, (long)T_ * HD_, (long)HD_ * T_, (long)T_ * T_, 0);
        softmax_kernel<<<BH * T_, 256, 0, stream>>>(s_f, a_bf);
        // AV: per (b,h): [T,T] @ [T,HD]
        gemm(a_bf, v_bf, nullptr, nullptr, nullptr, o_bf,
             T_, HD_, T_, BH, (long)T_ * T_, (long)T_ * HD_, (long)T_ * HD_, 0);
        repack_o_kernel<<<(ROWS * MD_) / 256, 256, 0, stream>>>(o_bf, o2_bf);
        gemm(o2_bf, wo_l, bo, h, h, nullptr, ROWS, MD_, MD_, 1, 0, 0, 0, 0);
        ln(h, ln2g, ln2b, y_bf, nullptr, nullptr, ROWS);
        gemm(y_bf, wf1_l, fb1, nullptr, nullptr, f_bf, ROWS, FF_, MD_, 1, 0, 0, 0, 1);
        gemm(f_bf, wf2_l, fb2, h, h, nullptr, ROWS, MD_, FF_, 1, 0, 0, 0, 0);
    }

    // ---- head + decoder ----
    ln(h, P(P_DYN_LNF_G), P(P_DYN_LNF_B), y_bf, nullptr, nullptr, ROWS);
    gemm(y_bf, w_out, P(P_DYN_OUT_B), nullptr, nullptr, pred_bf, ROWS, MD_, MD_, 1, 0, 0, 0, 0);
    gemm(pred_bf, w_d1, P(P_DEC_B1), nullptr, nullptr, d1_bf, ROWS, MD_, MD_, 1, 0, 0, 0, 1);
    gemm(d1_bf, w_d2, P(P_DEC_B2), nullptr, d2_f, nullptr, ROWS, MD_, MD_, 1, 0, 0, 0, 0);
    // final LN * latent_scale -> d_out
    ln(d2_f, P(P_OUT_LN_G), P(P_OUT_LN_B), nullptr, (float*)d_out, P(P_LATENT_SCALE), ROWS);
}
